// BasicBlockBase_5669356832550
// MI455X (gfx1250) — matrix operand order repaired and checked by reading, not compiled, not run on hardware
//
#include <hip/hip_runtime.h>
#include <hip/hip_bf16.h>

#define NPTS 100000
#define KOFF 27
#define CCH  64
#define EPSV 1e-5f
#define MT   4                      // 16-point subtiles per wave

typedef __bf16 bf16_t;
typedef __attribute__((ext_vector_type(16))) __bf16 v16bf;
typedef __attribute__((ext_vector_type(8)))  __bf16 v8bf;
typedef __attribute__((ext_vector_type(8)))  float  v8f;

static __device__ __forceinline__ bf16_t f2bf(float f) {
    union { float f; unsigned u; } x; x.f = f;
    unsigned r = (x.u + 0x7FFFu + ((x.u >> 16) & 1u)) >> 16;  // RNE
    union { unsigned short s; bf16_t b; } y; y.s = (unsigned short)r;
    return y.b;
}

// ---------- prep kernels ----------
__global__ void prep_feats_bf16(const float* __restrict__ f, bf16_t* __restrict__ fb, int n) {
    int i = blockIdx.x * blockDim.x + threadIdx.x;
    if (i < n) fb[i] = f2bf(f[i]);
}

// W[k][ci][co] fp32 -> Wt[k][co][ci] bf16  (B-fragment friendly: per-lane contiguous K run)
__global__ void prep_weights_bf16(const float* __restrict__ W, bf16_t* __restrict__ Wt) {
    int i = blockIdx.x * blockDim.x + threadIdx.x;
    if (i < KOFF * CCH * CCH) {
        int ci = i & 63;
        int co = (i >> 6) & 63;
        int k  = i >> 12;
        Wt[((size_t)k * CCH + co) * CCH + ci] = f2bf(W[((size_t)k * CCH + ci) * CCH + co]);
    }
}

// ---------- fused gather-GEMM (WMMA bf16) + BN stat accumulation ----------
// Each wave: MT x 16 points x 64 out-channels. B fragments loaded once per k
// and reused across the MT A-tiles (register-level weight reuse).
__global__ void __launch_bounds__(256)
gemm_gather_wmma(const bf16_t* __restrict__ src,
                 const bf16_t* __restrict__ Wt,
                 const int* __restrict__ nbr_idx,
                 const unsigned char* __restrict__ nbr_mask,
                 const float* __restrict__ bias,
                 float* __restrict__ out,
                 float* __restrict__ stats)
{
    __shared__ float sstats[2 * CCH];
    const int tid = threadIdx.x;
    if (tid < 2 * CCH) sstats[tid] = 0.0f;
    __syncthreads();

    const int lane = tid & 31;
    const int wave = tid >> 5;
    const int m    = lane & 15;   // point row within a 16-point subtile
    const int half = lane >> 4;   // lane half selects K sub-pattern
    const int tile0 = (blockIdx.x * 8 + wave) * MT;   // first 16-pt tile of this wave
    const int base0 = tile0 * 16;                     // first point of this wave

    v8f acc[MT][4] = {};
    const v16bf* __restrict__ Wv = (const v16bf*)Wt;

    // Per-subtile activity (wave-uniform; tiles are consecutive so act is monotone).
    bool act[MT];
    #pragma unroll
    for (int mt = 0; mt < MT; ++mt) act[mt] = ((tile0 + mt) * 16) < NPTS;

    if (act[0]) {
        // Single base pointers; mt offsets are compile-time immediates (16*KOFF*mt).
        const int* __restrict__ idxB = nbr_idx + (size_t)(base0 + m) * KOFF;
        const unsigned char* __restrict__ mskB = nbr_mask + (size_t)(base0 + m) * KOFF;

        for (int k = 0; k < KOFF; ++k) {
            // ---- B fragments for this k: 8 x v16bf, one clause of 16 b128 loads ----
            const size_t bb = ((size_t)k * CCH + m) * 4 + half;
            v16bf B0[4], B1[4];
            #pragma unroll
            for (int t = 0; t < 4; ++t) {
                B0[t] = Wv[bb + (size_t)t * 64];       // cin chunk [0,32)
                B1[t] = Wv[bb + (size_t)t * 64 + 2];   // cin chunk [32,64)
            }

            // ---- MT gathered A tiles, 8 WMMAs each ----
            #pragma unroll
            for (int mt = 0; mt < MT; ++mt) {
                if (!act[mt]) continue;                // wave-uniform
                const int nb = idxB[mt * 16 * KOFF + k];
                v16bf a0 = {}, a1 = {};
                if (mskB[mt * 16 * KOFF + k]) {
                    // A layout (matches B): lane half h, element e -> K = 16*h + e;
                    // two v8bf loads per chunk.
                    const v8bf* r = (const v8bf*)(src + (size_t)nb * CCH);
                    v8bf l0 = r[2 * half],     h0 = r[2 * half + 1];
                    v8bf l1 = r[2 * half + 4], h1 = r[2 * half + 5];
                    a0 = __builtin_shufflevector(l0, h0, 0,1,2,3,4,5,6,7,8,9,10,11,12,13,14,15);
                    a1 = __builtin_shufflevector(l1, h1, 0,1,2,3,4,5,6,7,8,9,10,11,12,13,14,15);
                }
                #pragma unroll
                for (int t = 0; t < 4; ++t)
                    acc[mt][t] = __builtin_amdgcn_wmma_f32_16x16x32_bf16(
                        false, a0, false, B0[t], (short)0, acc[mt][t], false, false);
                #pragma unroll
                for (int t = 0; t < 4; ++t)
                    acc[mt][t] = __builtin_amdgcn_wmma_f32_16x16x32_bf16(
                        false, a1, false, B1[t], (short)0, acc[mt][t], false, false);
            }
        }

        // ---- Epilogue: bias, store (coalesced per half-wave), per-channel stats ----
        #pragma unroll
        for (int mt = 0; mt < MT; ++mt) {
            if (!act[mt]) continue;
            const int tb = (tile0 + mt) * 16;
            #pragma unroll
            for (int t = 0; t < 4; ++t) {
                const int co = t * 16 + m;
                const float bv = bias ? bias[co] : 0.0f;
                float s = 0.0f, s2 = 0.0f;
                #pragma unroll
                for (int v = 0; v < 8; ++v) {          // C/D layout: VGPR v -> M = v + 8*half
                    const float x = acc[mt][t][v] + bv;
                    out[(size_t)(tb + v + 8 * half) * CCH + co] = x;
                    s += x; s2 += x * x;
                }
                atomicAdd(&sstats[co], s);
                atomicAdd(&sstats[CCH + co], s2);
            }
        }
    }
    __syncthreads();
    if (tid < 2 * CCH) atomicAdd(&stats[tid], sstats[tid]);
}

// ---------- BN finalize: scale/shift from accumulated stats ----------
__global__ void bn_finalize(const float* __restrict__ stats,
                            const float* __restrict__ gamma,
                            const float* __restrict__ beta,
                            float* __restrict__ ss)  // [128]: scale[64], shift[64]
{
    int c = threadIdx.x;
    if (c < CCH) {
        float mu  = stats[c] * (1.0f / NPTS);
        float var = stats[CCH + c] * (1.0f / NPTS) - mu * mu;
        float sc  = gamma[c] * rsqrtf(var + EPSV);
        ss[c]        = sc;
        ss[CCH + c]  = beta[c] - mu * sc;
    }
}

// ---------- BN apply + ReLU -> bf16 (feeds conv2 gather) ----------
__global__ void bn_relu_to_bf16(const float* __restrict__ x,
                                const float* __restrict__ ss,
                                bf16_t* __restrict__ h)
{
    int i = blockIdx.x * blockDim.x + threadIdx.x;
    if (i < NPTS * CCH) {
        int c = i & 63;
        float v = x[i] * ss[c] + ss[CCH + c];
        h[i] = f2bf(v > 0.0f ? v : 0.0f);
    }
}

// ---------- BN apply + residual + ReLU, in place on d_out ----------
__global__ void bn_residual_relu(float* __restrict__ x,
                                 const float* __restrict__ ss,
                                 const float* __restrict__ feats)
{
    int i = blockIdx.x * blockDim.x + threadIdx.x;
    if (i < NPTS * CCH) {
        int c = i & 63;
        float v = x[i] * ss[c] + ss[CCH + c] + feats[i];
        x[i] = v > 0.0f ? v : 0.0f;
    }
}

extern "C" void kernel_launch(void* const* d_in, const int* in_sizes, int n_in,
                              void* d_out, int out_size, void* d_ws, size_t ws_size,
                              hipStream_t stream) {
    const float*         feats    = (const float*)d_in[0];
    const float*         W1       = (const float*)d_in[1];
    const float*         b1       = (const float*)d_in[2];
    const float*         gamma1   = (const float*)d_in[3];
    const float*         beta1    = (const float*)d_in[4];
    const float*         W2       = (const float*)d_in[5];
    const float*         gamma2   = (const float*)d_in[6];
    const float*         beta2    = (const float*)d_in[7];
    const int*           nbr_idx  = (const int*)d_in[8];
    const unsigned char* nbr_mask = (const unsigned char*)d_in[9];
    float* out = (float*)d_out;

    char* ws = (char*)d_ws;
    size_t off = 0;
    auto alloc = [&](size_t bytes) -> void* {
        void* p = ws + off;
        off += (bytes + 255) & ~(size_t)255;
        return p;
    };
    bf16_t* featsBf = (bf16_t*)alloc((size_t)NPTS * CCH * sizeof(bf16_t));
    bf16_t* hBf     = (bf16_t*)alloc((size_t)NPTS * CCH * sizeof(bf16_t));
    bf16_t* Wt1     = (bf16_t*)alloc((size_t)KOFF * CCH * CCH * sizeof(bf16_t));
    bf16_t* Wt2     = (bf16_t*)alloc((size_t)KOFF * CCH * CCH * sizeof(bf16_t));
    float*  stats1  = (float*)alloc(2 * CCH * sizeof(float));
    float*  stats2  = (float*)alloc(2 * CCH * sizeof(float));
    float*  ss1     = (float*)alloc(2 * CCH * sizeof(float));
    float*  ss2     = (float*)alloc(2 * CCH * sizeof(float));

    hipMemsetAsync(stats1, 0, 2 * CCH * sizeof(float), stream);
    hipMemsetAsync(stats2, 0, 2 * CCH * sizeof(float), stream);

    const int nElem = NPTS * CCH;
    const int eb = (nElem + 255) / 256;
    const int wb = (KOFF * CCH * CCH + 255) / 256;
    prep_feats_bf16<<<eb, 256, 0, stream>>>(feats, featsBf, nElem);
    prep_weights_bf16<<<wb, 256, 0, stream>>>(W1, Wt1);
    prep_weights_bf16<<<wb, 256, 0, stream>>>(W2, Wt2);

    const int nTiles  = (NPTS + 15) / 16;             // 6250
    const int tilesPerBlock = 8 * MT;                 // 32
    const int nBlocks = (nTiles + tilesPerBlock - 1) / tilesPerBlock;  // 196

    // conv1 (+bias) with fused BN1 stats
    gemm_gather_wmma<<<nBlocks, 256, 0, stream>>>(featsBf, Wt1, nbr_idx, nbr_mask,
                                                  b1, out, stats1);
    bn_finalize<<<1, 64, 0, stream>>>(stats1, gamma1, beta1, ss1);
    bn_relu_to_bf16<<<eb, 256, 0, stream>>>(out, ss1, hBf);

    // conv2 (no bias) with fused BN2 stats
    gemm_gather_wmma<<<nBlocks, 256, 0, stream>>>(hBf, Wt2, nbr_idx, nbr_mask,
                                                  nullptr, out, stats2);
    bn_finalize<<<1, 64, 0, stream>>>(stats2, gamma2, beta2, ss2);
    bn_residual_relu<<<eb, 256, 0, stream>>>(out, ss2, feats);
}
